// HANLayer_67654324847387
// MI455X (gfx1250) — matrix operand order repaired and checked by reading, not compiled, not run on hardware
//
#include <hip/hip_runtime.h>
#include <hip/hip_bf16.h>
#include <math.h>

// ---------------- problem constants ----------------
constexpr int kN  = 50000;   // nodes
constexpr int kH  = 8;       // heads
constexpr int kD  = 32;      // per-head dim
constexpr int kE  = 400000;  // edges per meta-path
constexpr int kIN = 128;     // input features
constexpr int kHD = kH * kD; // 256
constexpr float kNegSlope = 0.2f;

// ---------------- WMMA types ----------------
typedef __attribute__((ext_vector_type(16))) _Float16 v16h;
typedef __attribute__((ext_vector_type(8)))  float    v8f;

__device__ __forceinline__ float leaky(float x) { return x > 0.f ? x : kNegSlope * x; }

// order-preserving float <-> uint encoding for atomicMax-based segment max
__device__ __forceinline__ unsigned fenc(float f) {
    unsigned u = __float_as_uint(f);
    return (u & 0x80000000u) ? ~u : (u | 0x80000000u);
}
__device__ __forceinline__ float fdec(unsigned k) {
    if (k == 0u) return 0.0f;  // empty segment -> reference maps -inf to 0
    unsigned u = (k & 0x80000000u) ? (k ^ 0x80000000u) : ~k;
    return __uint_as_float(u);
}

// ---------------- weight transpose + f32->f16 convert ----------------
// src: [K, Nc] row-major f32 ;  dst: [Nc, K] row-major f16
__global__ void wconv(const float* __restrict__ src, _Float16* __restrict__ dst,
                      int K, int Nc) {
    int t = blockIdx.x * blockDim.x + threadIdx.x;
    if (t >= K * Nc) return;
    int n = t / K, k = t - n * K;
    dst[t] = (_Float16)src[k * Nc + n];
}

__global__ void zero_f(float* __restrict__ p, int n) {
    int t = blockIdx.x * blockDim.x + threadIdx.x;
    if (t < n) p[t] = 0.f;
}

// out[n, c] = bias[c]
__global__ void set_bias(float* __restrict__ out, const float* __restrict__ bias) {
    int t = blockIdx.x * blockDim.x + threadIdx.x;
    if (t < kN * kHD) out[t] = bias[t & (kHD - 1)];
}

// ---------------- GEMM1: feat[N,256] = h[N,128] @ W[128,256]  (WMMA f16) ----------
// 512 threads = 16 waves; wave w computes 16x16 tile at cols [16w,16w+16), rows [16*blk,...)
__global__ __launch_bounds__(512) void gemm_feat(
    const float* __restrict__ A, const _Float16* __restrict__ bT, float* __restrict__ C) {
    const int lane = threadIdx.x & 31;
    const int wave = threadIdx.x >> 5;   // 0..15
    const int half = lane >> 4;          // K-half selector per ISA layout
    const int m15  = lane & 15;
    const float*    arow = A  + (size_t)(blockIdx.x * 16 + m15) * kIN;
    const _Float16* brow = bT + (size_t)(wave * 16 + m15) * kIN;
    v8f c = {};
#pragma unroll
    for (int ki = 0; ki < kIN / 32; ++ki) {
        const int k0 = ki * 32;
        const float4 a0 = *(const float4*)(arow + k0 + half * 8);
        const float4 a1 = *(const float4*)(arow + k0 + half * 8 + 4);
        const float4 a2 = *(const float4*)(arow + k0 + 16 + half * 8);
        const float4 a3 = *(const float4*)(arow + k0 + 16 + half * 8 + 4);
        v16h a;
        a[0]=(_Float16)a0.x; a[1]=(_Float16)a0.y; a[2]=(_Float16)a0.z; a[3]=(_Float16)a0.w;
        a[4]=(_Float16)a1.x; a[5]=(_Float16)a1.y; a[6]=(_Float16)a1.z; a[7]=(_Float16)a1.w;
        a[8]=(_Float16)a2.x; a[9]=(_Float16)a2.y; a[10]=(_Float16)a2.z; a[11]=(_Float16)a2.w;
        a[12]=(_Float16)a3.x; a[13]=(_Float16)a3.y; a[14]=(_Float16)a3.z; a[15]=(_Float16)a3.w;
        typedef __attribute__((ext_vector_type(8))) _Float16 v8h_;
        const v8h_ b0 = *(const v8h_*)(brow + k0 + half * 8);       // 16B aligned, table-order K
        const v8h_ b1 = *(const v8h_*)(brow + k0 + 16 + half * 8);
        v16h b = __builtin_shufflevector(b0, b1, 0,1,2,3,4,5,6,7,8,9,10,11,12,13,14,15);
        c = __builtin_amdgcn_wmma_f32_16x16x32_f16(false, a, false, b, (short)0, c, false, false);
    }
    float* crow = C + (size_t)(blockIdx.x * 16 + half * 8) * kHD + wave * 16 + m15;
#pragma unroll
    for (int r = 0; r < 8; ++r) crow[(size_t)r * kHD] = c[r];
}

// ---------------- per-(node,head) attention logits ----------------
__global__ void attn_logits(const float* __restrict__ feat,
                            const float* __restrict__ al, const float* __restrict__ ar,
                            float* __restrict__ el, float* __restrict__ er) {
    int t = blockIdx.x * blockDim.x + threadIdx.x;
    if (t >= kN * kH) return;
    int n = t >> 3, h = t & 7;
    const float* f = feat + (size_t)n * kHD + h * kD;
    const float* a = al + h * kD;
    const float* b = ar + h * kD;
    float sl = 0.f, sr = 0.f;
#pragma unroll
    for (int d = 0; d < kD; ++d) { float fv = f[d]; sl = fmaf(fv, a[d], sl); sr = fmaf(fv, b[d], sr); }
    el[t] = sl; er[t] = sr;
}

// ---------------- edge pass 1: per-dst segment max ----------------
__global__ void edge_max(const int* __restrict__ src, const int* __restrict__ dst,
                         const float* __restrict__ el, const float* __restrict__ er,
                         unsigned* __restrict__ mkey) {
    int t = blockIdx.x * blockDim.x + threadIdx.x;
    if (t >= kE * kH) return;
    int e = t >> 3, h = t & 7;
    int s = src[e], d = dst[e];
    float v = leaky(el[s * kH + h] + er[d * kH + h]);
    atomicMax(mkey + d * kH + h, fenc(v));
}

__global__ void decode_max(unsigned* __restrict__ mk, int n) {
    int t = blockIdx.x * blockDim.x + threadIdx.x;
    if (t < n) ((float*)mk)[t] = fdec(mk[t]);
}

// ---------------- edge pass 2: softmax denominator ----------------
__global__ void edge_den(const int* __restrict__ src, const int* __restrict__ dst,
                         const float* __restrict__ el, const float* __restrict__ er,
                         const float* __restrict__ m, float* __restrict__ den) {
    int t = blockIdx.x * blockDim.x + threadIdx.x;
    if (t >= kE * kH) return;
    int e = t >> 3, h = t & 7;
    int s = src[e], d = dst[e];
    float v = leaky(el[s * kH + h] + er[d * kH + h]);
    atomicAdd(den + d * kH + h, expf(v - m[d * kH + h]));
}

// ---------------- edge pass 3: weighted aggregation (1 wave / edge) ----------
__global__ __launch_bounds__(256) void edge_aggr(
    const int* __restrict__ src, const int* __restrict__ dst,
    const float* __restrict__ el, const float* __restrict__ er,
    const float* __restrict__ m, const float* __restrict__ den,
    const float* __restrict__ feat, float* __restrict__ out) {
    int wid = (blockIdx.x * blockDim.x + threadIdx.x) >> 5;
    if (wid >= kE) return;
    int lane = threadIdx.x & 31;
    int s = src[wid], d = dst[wid];
    float a = 0.f;
    if (lane < kH) {
        float v = leaky(el[s * kH + lane] + er[d * kH + lane]);
        a = expf(v - m[d * kH + lane]) / den[d * kH + lane];
    }
    const float* frow = feat + (size_t)s * kHD;
    float* orow = out + (size_t)d * kHD;
#pragma unroll
    for (int h = 0; h < kH; ++h) {
        float alpha = __shfl(a, h, 32);
        atomicAdd(orow + h * kD + lane, frow[h * kD + lane] * alpha);
    }
}

// ---------------- semantic MLP: sum_n tanh(z@sw1+sb1)@sw2  (WMMA f16) ---------
// 256 threads = 8 waves; wave w handles hidden cols [16w,16w+16), K=256
__global__ __launch_bounds__(256) void sem_reduce(
    const float* __restrict__ Z, const _Float16* __restrict__ swT,
    const float* __restrict__ sb1, const float* __restrict__ sw2,
    float* __restrict__ wsum) {
    const int lane = threadIdx.x & 31;
    const int wave = threadIdx.x >> 5;   // 0..7
    const int half = lane >> 4;
    const int m15  = lane & 15;
    const float*    arow = Z   + (size_t)(blockIdx.x * 16 + m15) * kHD;
    const _Float16* brow = swT + (size_t)(wave * 16 + m15) * kHD;
    v8f c = {};
#pragma unroll
    for (int ki = 0; ki < kHD / 32; ++ki) {
        const int k0 = ki * 32;
        const float4 a0 = *(const float4*)(arow + k0 + half * 8);
        const float4 a1 = *(const float4*)(arow + k0 + half * 8 + 4);
        const float4 a2 = *(const float4*)(arow + k0 + 16 + half * 8);
        const float4 a3 = *(const float4*)(arow + k0 + 16 + half * 8 + 4);
        v16h a;
        a[0]=(_Float16)a0.x; a[1]=(_Float16)a0.y; a[2]=(_Float16)a0.z; a[3]=(_Float16)a0.w;
        a[4]=(_Float16)a1.x; a[5]=(_Float16)a1.y; a[6]=(_Float16)a1.z; a[7]=(_Float16)a1.w;
        a[8]=(_Float16)a2.x; a[9]=(_Float16)a2.y; a[10]=(_Float16)a2.z; a[11]=(_Float16)a2.w;
        a[12]=(_Float16)a3.x; a[13]=(_Float16)a3.y; a[14]=(_Float16)a3.z; a[15]=(_Float16)a3.w;
        typedef __attribute__((ext_vector_type(8))) _Float16 v8h_;
        const v8h_ b0 = *(const v8h_*)(brow + k0 + half * 8);
        const v8h_ b1 = *(const v8h_*)(brow + k0 + 16 + half * 8);
        v16h b = __builtin_shufflevector(b0, b1, 0,1,2,3,4,5,6,7,8,9,10,11,12,13,14,15);
        c = __builtin_amdgcn_wmma_f32_16x16x32_f16(false, a, false, b, (short)0, c, false, false);
    }
    const int n = wave * 16 + m15;
    const float bias = sb1[n];
    const float w2   = sw2[n];
    float local = 0.f;
#pragma unroll
    for (int r = 0; r < 8; ++r) local += tanhf(c[r] + bias) * w2;
#pragma unroll
    for (int off = 16; off >= 1; off >>= 1) local += __shfl_xor(local, off, 32);
    if (lane == 0) atomicAdd(wsum, local);
}

// ---------------- meta-path softmax + combine ----------------
__global__ void beta_k(const float* __restrict__ wsum, float* __restrict__ beta) {
    float w0 = wsum[0] / (float)kN, w1 = wsum[1] / (float)kN;
    float mx = fmaxf(w0, w1);
    float e0 = expf(w0 - mx), e1 = expf(w1 - mx);
    float s = e0 + e1;
    beta[0] = e0 / s; beta[1] = e1 / s;
}

__global__ void combine(float* __restrict__ out, const float* __restrict__ z1,
                        const float* __restrict__ beta) {
    int t = blockIdx.x * blockDim.x + threadIdx.x;
    if (t < kN * kHD) out[t] = beta[0] * out[t] + beta[1] * z1[t];
}

// ---------------- host launcher ----------------
extern "C" void kernel_launch(void* const* d_in, const int* in_sizes, int n_in,
                              void* d_out, int out_size, void* d_ws, size_t ws_size,
                              hipStream_t stream) {
    (void)in_sizes; (void)n_in; (void)out_size; (void)ws_size;
    const float* h   = (const float*)d_in[0];
    const int*   e0  = (const int*)  d_in[1];
    const int*   e1  = (const int*)  d_in[2];
    const float* fc0 = (const float*)d_in[3];
    const float* al0 = (const float*)d_in[4];
    const float* ar0 = (const float*)d_in[5];
    const float* b0  = (const float*)d_in[6];
    const float* fc1 = (const float*)d_in[7];
    const float* al1 = (const float*)d_in[8];
    const float* ar1 = (const float*)d_in[9];
    const float* b1  = (const float*)d_in[10];
    const float* sw1 = (const float*)d_in[11];
    const float* sb1 = (const float*)d_in[12];
    const float* sw2 = (const float*)d_in[13];
    float* out = (float*)d_out;

    // workspace layout: f16 weight copies first (keeps 16B alignment for v8h loads)
    _Float16* bT0 = (_Float16*)d_ws;          // [256,128] f16
    _Float16* bT1 = bT0 + kIN * kHD;          // [256,128] f16
    _Float16* swT = bT1 + kIN * kHD;          // [128,256] f16
    float* feat = (float*)((char*)d_ws + 3 * (size_t)kIN * kHD * sizeof(_Float16));
    float* z1   = feat + (size_t)kN * kHD;    // z of meta-path 1 (z0 lives in d_out)
    float* el   = z1 + (size_t)kN * kHD;
    float* er   = el + kN * kH;
    float* mbuf = er + kN * kH;               // also aliased as unsigned keys
    float* den  = mbuf + kN * kH;             // contiguous after mbuf (zeroed together)
    float* wsum = den + kN * kH;              // [2]
    float* beta = wsum + 2;                   // [2]

    // one-time (per launch) weight conversion
    wconv<<<dim3((kIN * kHD + 255) / 256), dim3(256), 0, stream>>>(fc0, bT0, kIN, kHD);
    wconv<<<dim3((kIN * kHD + 255) / 256), dim3(256), 0, stream>>>(fc1, bT1, kIN, kHD);
    wconv<<<dim3((kHD * kIN + 255) / 256), dim3(256), 0, stream>>>(sw1, swT, kHD, kIN);
    zero_f<<<dim3(1), dim3(32), 0, stream>>>(wsum, 2);

    struct Path { const int* edges; const _Float16* bT; const float* al; const float* ar;
                  const float* bias; float* outbuf; };
    Path P[2] = { { e0, bT0, al0, ar0, b0, out },
                  { e1, bT1, al1, ar1, b1, z1  } };

    for (int p = 0; p < 2; ++p) {
        const int* src = P[p].edges;
        const int* dst = P[p].edges + kE;
        gemm_feat<<<dim3(kN / 16), dim3(512), 0, stream>>>(h, P[p].bT, feat);
        attn_logits<<<dim3((kN * kH + 255) / 256), dim3(256), 0, stream>>>(
            feat, P[p].al, P[p].ar, el, er);
        zero_f<<<dim3((2 * kN * kH + 255) / 256), dim3(256), 0, stream>>>(mbuf, 2 * kN * kH);
        edge_max<<<dim3((kE * kH + 255) / 256), dim3(256), 0, stream>>>(
            src, dst, el, er, (unsigned*)mbuf);
        decode_max<<<dim3((kN * kH + 255) / 256), dim3(256), 0, stream>>>(
            (unsigned*)mbuf, kN * kH);
        edge_den<<<dim3((kE * kH + 255) / 256), dim3(256), 0, stream>>>(
            src, dst, el, er, mbuf, den);
        set_bias<<<dim3((kN * kHD + 255) / 256), dim3(256), 0, stream>>>(
            P[p].outbuf, P[p].bias);
        edge_aggr<<<dim3(kE / 8), dim3(256), 0, stream>>>(
            src, dst, el, er, mbuf, den, feat, P[p].outbuf);
        sem_reduce<<<dim3(kN / 16), dim3(256), 0, stream>>>(
            P[p].outbuf, swT, sb1, sw2, wsum + p);
    }

    beta_k<<<dim3(1), dim3(1), 0, stream>>>(wsum, beta);
    combine<<<dim3((kN * kHD + 255) / 256), dim3(256), 0, stream>>>(out, z1, beta);
}
